// ExpertParallelLayer_16372415333091
// MI455X (gfx1250) — matrix operand order repaired and checked by reading, not compiled, not run on hardware
//
#include <hip/hip_runtime.h>
#include <hip/hip_bf16.h>

#define D_DIM 2048
#define B_TOK 2048
#define E_NUM 16
#define BLK_M 64
#define BLK_N 64
#define BLK_K 32
#define MT_MAX (B_TOK / BLK_M)   // 32 (worst case: all tokens on one expert)
#define NT (D_DIM / BLK_N)       // 32

typedef __attribute__((ext_vector_type(16))) _Float16 v16h;
typedef __attribute__((ext_vector_type(8)))  _Float16 h8;
typedef __attribute__((ext_vector_type(4)))  _Float16 h4;
typedef __attribute__((ext_vector_type(8)))  float    v8f;
typedef __attribute__((ext_vector_type(4)))  float    f4v;

#define SHUF16(LO, HI) __builtin_shufflevector((LO), (HI), 0,1,2,3,4,5,6,7,8,9,10,11,12,13,14,15)

// ---------------------------------------------------------------- init ------
__global__ void init_kernel(float* out, size_t n) {
  size_t i = (size_t)blockIdx.x * blockDim.x + threadIdx.x;
  size_t stride = (size_t)gridDim.x * blockDim.x;
  for (; i < n; i += stride) out[i] = 0.0f;
}

// ---------------------------------------------------------------- gating ----
__global__ __launch_bounds__(128) void gate_kernel(
    const float* __restrict__ x, const float* __restrict__ Wg,
    const float* __restrict__ bg,
    int* __restrict__ top1, int* __restrict__ top2,
    float* __restrict__ w1o, float* __restrict__ w2o,
    float* __restrict__ impTok)
{
  const int b = blockIdx.x * 4 + (threadIdx.x >> 5);
  const int lane = threadIdx.x & 31;
  float acc[E_NUM];
#pragma unroll
  for (int e = 0; e < E_NUM; ++e) acc[e] = 0.0f;
  const float* xr = x + (size_t)b * D_DIM;
  for (int d = lane; d < D_DIM; d += 32) {
    float xv = xr[d];
    const float* wr = Wg + (size_t)d * E_NUM;
#pragma unroll
    for (int e = 0; e < E_NUM; ++e) acc[e] += xv * wr[e];
  }
#pragma unroll
  for (int off = 16; off >= 1; off >>= 1) {
#pragma unroll
    for (int e = 0; e < E_NUM; ++e) acc[e] += __shfl_xor(acc[e], off, 32);
  }
  float s[E_NUM];
#pragma unroll
  for (int e = 0; e < E_NUM; ++e) s[e] = acc[e] + bg[e];

  int i1 = 0; float m1 = s[0];
#pragma unroll
  for (int e = 1; e < E_NUM; ++e) if (s[e] > m1) { m1 = s[e]; i1 = e; }
  int i2 = (i1 == 0) ? 1 : 0; float m2 = s[i2];
#pragma unroll
  for (int e = 0; e < E_NUM; ++e) if (e != i1 && s[e] > m2) { m2 = s[e]; i2 = e; }

  float wA = 1.0f / (1.0f + __expf(m2 - m1));   // softmax over {m1,m2}
  float wB = 1.0f - wA;

  float ssum = 0.0f;
#pragma unroll
  for (int e = 0; e < E_NUM; ++e) ssum += __expf(s[e] - m1);
  if (lane < E_NUM) impTok[(size_t)b * E_NUM + lane] = __expf(s[lane] - m1) / ssum;
  if (lane == 0) { top1[b] = i1; top2[b] = i2; w1o[b] = wA; w2o[b] = wB; }
}

// ---------------------------------------------------------------- routing ---
__global__ __launch_bounds__(32) void route_kernel(
    const int* __restrict__ top1, const int* __restrict__ top2,
    const float* __restrict__ w1, const float* __restrict__ w2,
    int* __restrict__ cnt, int* __restrict__ toklist, float* __restrict__ wlist)
{
  const int e = blockIdx.x;
  const int lane = threadIdx.x;
  int pos = 0;
  for (int b0 = 0; b0 < B_TOK; b0 += 32) {
    int b = b0 + lane;
    bool hit1 = (top1[b] == e);
    bool hit  = hit1 || (top2[b] == e);
    unsigned long long mask = __ballot(hit);
    int pre = __popcll(mask & ((1ull << lane) - 1ull));
    if (hit) {
      toklist[e * B_TOK + pos + pre] = b;
      wlist[e * B_TOK + pos + pre]   = hit1 ? w1[b] : w2[b];
    }
    pos += __popcll(mask & 0xffffffffull);
  }
  if (lane == 0) cnt[e] = pos;
}

// ---------------------------------------------------------------- losses ----
__global__ __launch_bounds__(256) void finalize_kernel(
    const int* __restrict__ cnt, int* __restrict__ offs,
    const float* __restrict__ impTok, float* __restrict__ losses)
{
  __shared__ float part[256];
  __shared__ float imp[E_NUM];
  const int t = threadIdx.x;
  const int e = t & 15;
  const int g = t >> 4;
  float sum = 0.0f;
  for (int b = g; b < B_TOK; b += 16) sum += impTok[(size_t)b * E_NUM + e];
  part[t] = sum;
  __syncthreads();
  if (t < E_NUM) {
    float s = 0.0f;
    for (int g2 = 0; g2 < 16; ++g2) s += part[g2 * 16 + t];
    imp[t] = s;
  }
  __syncthreads();
  if (t == 0) {
    int o = 0; float meanL = 0.0f; int c[E_NUM];
    for (int i = 0; i < E_NUM; ++i) { c[i] = cnt[i]; offs[i] = o; o += c[i]; meanL += (float)c[i]; }
    meanL /= (float)E_NUM;
    float varL = 0.0f;
    for (int i = 0; i < E_NUM; ++i) { float d = (float)c[i] - meanL; varL += d * d; }
    varL /= (float)(E_NUM - 1);                       // ddof=1
    losses[0] = varL / (float)(E_NUM * (B_TOK / E_NUM));
    float mi = 0.0f;
    for (int i = 0; i < E_NUM; ++i) mi += imp[i];
    mi /= (float)E_NUM;
    float vi = 0.0f;
    for (int i = 0; i < E_NUM; ++i) { float d = imp[i] - mi; vi += d * d; }
    vi /= (float)(E_NUM - 1);
    losses[1] = vi / (mi + 1e-8f);
  }
}

// ---------------------------------------------------------------- GEMM 1 ----
// H = relu(gather(x) @ W1[e] + b1[e]); 64x64 tile, 8 waves, WMMA f16.
__global__ __launch_bounds__(256) void gemm1_kernel(
    const float* __restrict__ x, const float* __restrict__ W1,
    const float* __restrict__ b1,
    const int* __restrict__ cnt, const int* __restrict__ offs,
    const int* __restrict__ toklist, _Float16* __restrict__ H)
{
  const int e  = blockIdx.y >> 5;
  const int mt = blockIdx.y & 31;
  const int ce = cnt[e];
  const int mbase = mt * BLK_M;
  if (mbase >= ce) return;
  const int nbase = blockIdx.x * BLK_N;

  __shared__ _Float16 As[BLK_M][BLK_K + 8];     // row-major in K, 80B pitch
  __shared__ _Float16 Bs_t[BLK_N][BLK_K + 8];   // transposed: [n][k]

  const int t = threadIdx.x;
  const int lane = t & 31;
  const int w  = t >> 5;
  const int m0 = (w & 3) * 16;
  const int n0 = (w >> 2) * 32;
  const int hi = lane >> 4;
  const int l15 = lane & 15;

  v8f c0 = {}; v8f c1 = {};
  const float* Wbase = W1 + (size_t)e * D_DIM * D_DIM;
  const int* tl = toklist + e * B_TOK;

  // Hoisted gather state: clamped row pointer + 0/1 scale (uniform flow).
  const float* aptr[2]; float ascale[2]; int ar[2], akc[2];
#pragma unroll
  for (int i = 0; i < 2; ++i) {
    int v = t + i * 256;             // 512 float4 chunks of the 64x32 A tile
    int r = v >> 3, kc = (v & 7) * 4;
    int mg = mbase + r;
    int mgc = (mg < ce) ? mg : (ce - 1);
    aptr[i]  = x + (size_t)tl[mgc] * D_DIM + kc;
    ascale[i] = (mg < ce) ? 1.0f : 0.0f;
    ar[i] = r; akc[i] = kc;
  }

  // k-invariant LDS fragment pointers (all 16B aligned, ds_load_b128)
  const h8* aLo  = (const h8*)&As[m0 + l15][hi * 16];
  const h8* aHi  = (const h8*)&As[m0 + l15][hi * 16 + 8];
  const h8* b0Lo = (const h8*)&Bs_t[n0 + l15][hi * 16];
  const h8* b0Hi = (const h8*)&Bs_t[n0 + l15][hi * 16 + 8];
  const h8* b1Lo = (const h8*)&Bs_t[n0 + 16 + l15][hi * 16];
  const h8* b1Hi = (const h8*)&Bs_t[n0 + 16 + l15][hi * 16 + 8];

  for (int k0 = 0; k0 < D_DIM; k0 += BLK_K) {
#pragma unroll
    for (int i = 0; i < 2; ++i) {               // A: x rows f32x4 -> f16x4
      f4v fv = *(const f4v*)(aptr[i] + k0);
      *(h4*)&As[ar[i]][akc[i]] = __builtin_convertvector(fv * ascale[i], h4);
    }
#pragma unroll
    for (int i = 0; i < 2; ++i) {               // B: W slab f32x4 -> transposed f16
      int v = t + i * 256;
      int kk = v >> 4, nc = (v & 15) * 4;
      f4v wv = *(const f4v*)&Wbase[(size_t)(k0 + kk) * D_DIM + nbase + nc];
      h4 hv = __builtin_convertvector(wv, h4);
      Bs_t[nc + 0][kk] = hv[0]; Bs_t[nc + 1][kk] = hv[1];
      Bs_t[nc + 2][kk] = hv[2]; Bs_t[nc + 3][kk] = hv[3];
    }
    if (k0 + BLK_K < D_DIM)
      __builtin_prefetch(&Wbase[(size_t)(k0 + BLK_K) * D_DIM + nbase + (t & 63)], 0, 1);
    __syncthreads();

    v16h a   = SHUF16(*aLo, *aHi);
    v16h bv0 = SHUF16(*b0Lo, *b0Hi);
    v16h bv1 = SHUF16(*b1Lo, *b1Hi);
    c0 = __builtin_amdgcn_wmma_f32_16x16x32_f16(false, a, false, bv0, (short)0, c0, false, false);
    c1 = __builtin_amdgcn_wmma_f32_16x16x32_f16(false, a, false, bv1, (short)0, c1, false, false);
    __syncthreads();
  }

  const float* bias = b1 + (size_t)e * D_DIM;
  const int rb = offs[e];
#pragma unroll
  for (int r = 0; r < 8; ++r) {                 // C layout: M = r + 8*hi, N = lane&15
    int mg = mbase + m0 + hi * 8 + r;
    if (mg < ce) {
      int ng0 = nbase + n0 + l15;
      int ng1 = ng0 + 16;
      float v0 = c0[r] + bias[ng0]; v0 = v0 > 0.0f ? v0 : 0.0f;
      float v1 = c1[r] + bias[ng1]; v1 = v1 > 0.0f ? v1 : 0.0f;
      _Float16* hr = H + (size_t)(rb + mg) * D_DIM;
      hr[ng0] = (_Float16)v0;
      hr[ng1] = (_Float16)v1;
    }
  }
}

// ---------------------------------------------------------------- GEMM 2 ----
// out[tok] += w * (H @ W2[e] + b2[e]); A tile staged via async global->LDS.
__global__ __launch_bounds__(256) void gemm2_kernel(
    const _Float16* __restrict__ H, const float* __restrict__ W2,
    const float* __restrict__ b2,
    const int* __restrict__ cnt, const int* __restrict__ offs,
    const int* __restrict__ toklist, const float* __restrict__ wlist,
    float* __restrict__ out)
{
  const int e  = blockIdx.y >> 5;
  const int mt = blockIdx.y & 31;
  const int ce = cnt[e];
  const int mbase = mt * BLK_M;
  if (mbase >= ce) return;
  const int nbase = blockIdx.x * BLK_N;

  __shared__ _Float16 As[BLK_M][BLK_K + 8];
  __shared__ _Float16 Bs_t[BLK_N][BLK_K + 8];

  const int t = threadIdx.x;
  const int lane = t & 31;
  const int w  = t >> 5;
  const int m0 = (w & 3) * 16;
  const int n0 = (w >> 2) * 32;
  const int hi = lane >> 4;
  const int l15 = lane & 15;

  v8f c0 = {}; v8f c1 = {};
  const float* Wbase = W2 + (size_t)e * D_DIM * D_DIM;
  const int rb = offs[e];

  // Hoisted async-copy state: per-thread 8-byte chunk of the 64x32 f16 tile.
  unsigned ldsoff[2]; const _Float16* hrow[2]; bool avalid[2]; int ar[2], akc[2];
#pragma unroll
  for (int i = 0; i < 2; ++i) {
    int v = t + i * 256;             // 512 b64 chunks
    int r = v >> 3, kc = (v & 7) * 4;
    int mg = mbase + r;
    int mgc = (mg < ce) ? mg : (ce - 1);
    ldsoff[i] = (unsigned)(size_t)&As[r][kc];
    hrow[i]   = H + (size_t)(rb + mgc) * D_DIM + kc;
    avalid[i] = (mg < ce);
    ar[i] = r; akc[i] = kc;
  }

  const h8* aLo  = (const h8*)&As[m0 + l15][hi * 16];
  const h8* aHi  = (const h8*)&As[m0 + l15][hi * 16 + 8];
  const h8* b0Lo = (const h8*)&Bs_t[n0 + l15][hi * 16];
  const h8* b0Hi = (const h8*)&Bs_t[n0 + l15][hi * 16 + 8];
  const h8* b1Lo = (const h8*)&Bs_t[n0 + 16 + l15][hi * 16];
  const h8* b1Hi = (const h8*)&Bs_t[n0 + 16 + l15][hi * 16 + 8];

  for (int k0 = 0; k0 < D_DIM; k0 += BLK_K) {
    // A: H rows are already f16 -> async DMA straight into LDS (ASYNCcnt).
#pragma unroll
    for (int i = 0; i < 2; ++i) {
      unsigned long long ga = (unsigned long long)(size_t)(hrow[i] + k0);
      asm volatile("global_load_async_to_lds_b64 %0, %1, off"
                   :: "v"(ldsoff[i]), "v"(ga) : "memory");
    }
#pragma unroll
    for (int i = 0; i < 2; ++i) {               // B: W2 slab f32x4 -> transposed f16
      int v = t + i * 256;
      int kk = v >> 4, nc = (v & 15) * 4;
      f4v wv = *(const f4v*)&Wbase[(size_t)(k0 + kk) * D_DIM + nbase + nc];
      h4 hv = __builtin_convertvector(wv, h4);
      Bs_t[nc + 0][kk] = hv[0]; Bs_t[nc + 1][kk] = hv[1];
      Bs_t[nc + 2][kk] = hv[2]; Bs_t[nc + 3][kk] = hv[3];
    }
    if (k0 + BLK_K < D_DIM)
      __builtin_prefetch(&Wbase[(size_t)(k0 + BLK_K) * D_DIM + nbase + (t & 63)], 0, 1);

    asm volatile("s_wait_asynccnt 0x0" ::: "memory");
#pragma unroll
    for (int i = 0; i < 2; ++i)                 // zero the padding rows
      if (!avalid[i]) { h4 z = {}; *(h4*)&As[ar[i]][akc[i]] = z; }
    __syncthreads();

    v16h a   = SHUF16(*aLo, *aHi);
    v16h bv0 = SHUF16(*b0Lo, *b0Hi);
    v16h bv1 = SHUF16(*b1Lo, *b1Hi);
    c0 = __builtin_amdgcn_wmma_f32_16x16x32_f16(false, a, false, bv0, (short)0, c0, false, false);
    c1 = __builtin_amdgcn_wmma_f32_16x16x32_f16(false, a, false, bv1, (short)0, c1, false, false);
    __syncthreads();
  }

  const float* bias = b2 + (size_t)e * D_DIM;
#pragma unroll
  for (int r = 0; r < 8; ++r) {
    int mg = mbase + m0 + hi * 8 + r;
    if (mg < ce) {
      int tok  = toklist[e * B_TOK + mg];
      float wt = wlist[e * B_TOK + mg];
      int ng0 = nbase + n0 + l15;
      int ng1 = ng0 + 16;
      float y0 = c0[r] + bias[ng0];
      float y1 = c1[r] + bias[ng1];
      atomicAdd(&out[(size_t)tok * D_DIM + ng0], wt * y0);
      atomicAdd(&out[(size_t)tok * D_DIM + ng1], wt * y1);
    }
  }
}

// ---------------------------------------------------------------- launch ----
extern "C" void kernel_launch(void* const* d_in, const int* in_sizes, int n_in,
                              void* d_out, int out_size, void* d_ws, size_t ws_size,
                              hipStream_t stream) {
  (void)in_sizes; (void)n_in; (void)ws_size;
  const float* x  = (const float*)d_in[0];
  const float* W1 = (const float*)d_in[1];
  const float* b1 = (const float*)d_in[2];
  const float* W2 = (const float*)d_in[3];
  const float* b2 = (const float*)d_in[4];
  const float* Wg = (const float*)d_in[5];
  const float* bg = (const float*)d_in[6];
  float* out = (float*)d_out;

  char* p = (char*)d_ws;
  int*   cnt    = (int*)p;    p += 256;
  int*   offs   = (int*)p;    p += 256;
  int*   top1   = (int*)p;    p += B_TOK * 4;
  int*   top2   = (int*)p;    p += B_TOK * 4;
  float* w1v    = (float*)p;  p += B_TOK * 4;
  float* w2v    = (float*)p;  p += B_TOK * 4;
  float* impTok = (float*)p;  p += (size_t)B_TOK * E_NUM * 4;
  int*   tokl   = (int*)p;    p += (size_t)E_NUM * B_TOK * 4;
  float* wl     = (float*)p;  p += (size_t)E_NUM * B_TOK * 4;
  _Float16* H   = (_Float16*)p;     // B*K*D f16 = 16 MB

  init_kernel<<<2048, 256, 0, stream>>>(out, (size_t)out_size);
  gate_kernel<<<B_TOK / 4, 128, 0, stream>>>(x, Wg, bg, top1, top2, w1v, w2v, impTok);
  route_kernel<<<E_NUM, 32, 0, stream>>>(top1, top2, w1v, w2v, cnt, tokl, wl);
  finalize_kernel<<<1, 256, 0, stream>>>(cnt, offs, impTok, out + (size_t)B_TOK * D_DIM);

  dim3 g(NT, E_NUM * MT_MAX);
  gemm1_kernel<<<g, 256, 0, stream>>>(x, W1, b1, cnt, offs, tokl, H);
  gemm2_kernel<<<g, 256, 0, stream>>>(H, W2, b2, cnt, offs, tokl, wl, out);
}
